// ModelS_35802847380146
// MI455X (gfx1250) — matrix operand order repaired and checked by reading, not compiled, not run on hardware
//
#include <hip/hip_runtime.h>

// ---------------- problem constants ----------------
#define GG 4
#define NN 256
#define HH 256
#define EUD 32640            // N*(N-1)/2
#define MNODE (GG * NN)      // 1024
#define COEF (1.0f / 255.0f) // complete graph: deg = N-1 = 255 -> dinv^2 = 1/255
#define SLOPE 0.1f

// ---------------- GEMM tiling ----------------
#define BM 128
#define BN 128
#define BK 32

typedef __attribute__((ext_vector_type(16))) __bf16 v16bf;
typedef __attribute__((ext_vector_type(8)))  __bf16 v8bf;
typedef __attribute__((ext_vector_type(8)))  float  v8f;
typedef __attribute__((ext_vector_type(8)))  unsigned short u16x8;

__device__ __forceinline__ float bf2f(unsigned short u) {
  union { unsigned int i; float f; } c; c.i = ((unsigned int)u) << 16; return c.f;
}
__device__ __forceinline__ unsigned short f2bf(float f) {
  union { float f; unsigned int i; } c; c.f = f;
  unsigned int u = c.i;
  u += 0x7FFFu + ((u >> 16) & 1u);   // round-to-nearest-even
  return (unsigned short)(u >> 16);
}

union Frag16 { v16bf v; unsigned short u[16]; u16x8 h[2]; };

// ---------------------------------------------------------------------------
// Tiled bf16 WMMA GEMM:  C(MxNc) = act( A(MxK) @ B(KxNc) + bias )
//  - A: row-major bf16, or gathered on the fly: row r -> xg[base+ud[2r]] + xg[base+ud[2r+1]]
//    (gather sum uses packed bf16 adds -> v_pk_add_bf16)
//  - B: row-major bf16 (K x Nc)
//  - outputs: f32 (Cf) and/or bf16 (Cb), leading dim Nc
// 256 threads = 8 waves; waves tile 4x2 over (M,N); each wave owns a 32x64
// patch = 2x4 wmma tiles -> 8 v_wmma_f32_16x16x32_bf16 per k-step.
// Double-buffered LDS: one barrier per k-step, global prefetch overlaps wmma.
// ---------------------------------------------------------------------------
__global__ __launch_bounds__(256)
void gemm_bf16(const unsigned short* __restrict__ A,
               const unsigned short* __restrict__ XgB,
               const int* __restrict__ ud, int node_base,
               const unsigned short* __restrict__ B,
               const float* __restrict__ bias,
               float* __restrict__ Cf,
               unsigned short* __restrict__ Cb,
               int M, int Nc, int K, int gatherA, int leaky)
{
  __shared__ __align__(16) unsigned short sA[2][BM * BK];   // 2 x 8 KB
  __shared__ __align__(16) unsigned short sBt[2][BN * BK];  // 2 x 8 KB, [n][k]

  const int tid    = threadIdx.x;
  const int lane   = tid & 31;
  const int wave   = tid >> 5;
  const int wm     = (wave & 3) << 5;   // wave M offset in tile (4 waves * 32 = 128)
  const int wn     = (wave >> 2) << 6;  // wave N offset in tile (2 waves * 64 = 128)
  const int lane15 = lane & 15;
  const int laneHi = lane >> 4;
  const int tileM  = blockIdx.y * BM;
  const int tileN  = blockIdx.x * BN;

  // ---- per-thread staging coordinates (k-invariant) ----
  // A: each thread stages 16 contiguous k of one row: row = tid/2, k0off = (tid&1)*16
  const int ar  = tid >> 1;
  const int ac0 = (tid & 1) << 4;
  const int agr = tileM + ar;
  const bool arow_ok = (agr < M);
  const unsigned short* aprow0 = nullptr;
  const unsigned short* aprow1 = nullptr;
  if (arow_ok) {
    if (!gatherA) {
      aprow0 = A + (size_t)agr * K;
    } else {
      int u0 = ud[2 * agr], u1 = ud[2 * agr + 1];      // hoisted: one b64 load total
      aprow0 = XgB + (size_t)(node_base + u0) * K;
      aprow1 = XgB + (size_t)(node_base + u1) * K;
    }
  }
  // B: each thread stages 16 contiguous n of one k-row: krow = tid/8, n0 = (tid&7)*16
  const int br  = tid >> 3;
  const int bc0 = (tid & 7) << 4;
  const bool bvec_ok = (tileN + bc0 + 15) < Nc;        // whole 16-vector in range

  // stage tile kt from global into registers
  auto stage = [&](int kt, u16x8& a0, u16x8& a1, u16x8& b0, u16x8& b1) {
    const int k0 = kt << 5;
    a0 = {}; a1 = {};
    if (arow_ok) {
      const u16x8* p = (const u16x8*)(aprow0 + k0 + ac0);
      a0 = p[0]; a1 = p[1];
      if (gatherA) {
        const u16x8* q = (const u16x8*)(aprow1 + k0 + ac0);
        v8bf s0 = __builtin_bit_cast(v8bf, a0) + __builtin_bit_cast(v8bf, q[0]);
        v8bf s1 = __builtin_bit_cast(v8bf, a1) + __builtin_bit_cast(v8bf, q[1]);
        a0 = __builtin_bit_cast(u16x8, s0);
        a1 = __builtin_bit_cast(u16x8, s1);
      }
    }
    b0 = {}; b1 = {};
    if (bvec_ok) {
      const u16x8* p = (const u16x8*)&B[(size_t)(k0 + br) * Nc + tileN + bc0];
      b0 = p[0]; b1 = p[1];
    } else {
#pragma unroll
      for (int e = 0; e < 8; ++e) {
        int gn0 = tileN + bc0 + e, gn1 = gn0 + 8;
        b0[e] = (gn0 < Nc) ? B[(size_t)(k0 + br) * Nc + gn0] : (unsigned short)0;
        b1[e] = (gn1 < Nc) ? B[(size_t)(k0 + br) * Nc + gn1] : (unsigned short)0;
      }
    }
  };
  // commit staged registers into LDS buffer `buf`
  auto commit = [&](int buf, u16x8 a0, u16x8 a1, u16x8 b0, u16x8 b1) {
    unsigned short* sa = &sA[buf][0];
    unsigned short* sb = &sBt[buf][0];
    *(u16x8*)&sa[ar * BK + ac0]     = a0;
    *(u16x8*)&sa[ar * BK + ac0 + 8] = a1;
#pragma unroll
    for (int e = 0; e < 8; ++e) {
      sb[(bc0 + e) * BK + br]     = b0[e];
      sb[(bc0 + 8 + e) * BK + br] = b1[e];
    }
  };

  v8f acc[2][4] = {};
  const int nk = K >> 5;

  u16x8 pa0, pa1, pb0, pb1;
  stage(0, pa0, pa1, pb0, pb1);
  commit(0, pa0, pa1, pb0, pb1);
  __syncthreads();

  for (int kt = 0; kt < nk; ++kt) {
    const int cur = kt & 1;
    const bool more = (kt + 1) < nk;
    if (more) stage(kt + 1, pa0, pa1, pb0, pb1);   // global prefetch overlaps wmma

    // ---- fragments per CDNA5 16-bit A/B VGPR layouts ----
    // A (16x32): lanes 0-15 -> M=lane, elems[0..7]=K 0..7, elems[8..15]=K 16..23
    //            lanes 16-31 -> K 8..15 / 24..31
    const unsigned short* sa = &sA[cur][0];
    const unsigned short* sb = &sBt[cur][0];
    Frag16 af[2], bfg[4];
#pragma unroll
    for (int mi = 0; mi < 2; ++mi) {
      const unsigned short* p = &sa[(wm + mi * 16 + lane15) * BK + laneHi * 8];
      af[mi].h[0] = *(const u16x8*)p;          // K run 0
      af[mi].h[1] = *(const u16x8*)(p + 16);   // K run +16
    }
    // B (32x16): lanes 0-15 -> N=lane, K 0..7 / 16..23; lanes 16-31 -> K 8..15 / 24..31
#pragma unroll
    for (int ni = 0; ni < 4; ++ni) {
      const unsigned short* p = &sb[(wn + ni * 16 + lane15) * BK + laneHi * 8];
      bfg[ni].h[0] = *(const u16x8*)p;
      bfg[ni].h[1] = *(const u16x8*)(p + 16);
    }

#pragma unroll
    for (int mi = 0; mi < 2; ++mi)
#pragma unroll
      for (int ni = 0; ni < 4; ++ni)
        acc[mi][ni] = __builtin_amdgcn_wmma_f32_16x16x32_bf16(
            false, af[mi].v, false, bfg[ni].v, (short)0, acc[mi][ni], false, false);

    if (more) {
      commit(1 - cur, pa0, pa1, pb0, pb1);
      __syncthreads();
    }
  }

  // ---- epilogue: D layout -> row = base + r + (laneHi?8:0), col = base + lane15 ----
#pragma unroll
  for (int mi = 0; mi < 2; ++mi)
#pragma unroll
    for (int ni = 0; ni < 4; ++ni) {
      int col = tileN + wn + ni * 16 + lane15;
      if (col >= Nc) continue;
      float bv = bias ? bias[col] : 0.0f;
#pragma unroll
      for (int r = 0; r < 8; ++r) {
        int row = tileM + wm + mi * 16 + laneHi * 8 + r;
        if (row >= M) continue;
        float v = acc[mi][ni][r] + bv;
        if (leaky) v = (v >= 0.0f) ? v : SLOPE * v;
        if (Cf) Cf[(size_t)row * Nc + col] = v;
        if (Cb) Cb[(size_t)row * Nc + col] = f2bf(v);
      }
    }
}

// ---------------- small helper kernels ----------------

// f32 (KxN) -> zero-padded bf16 (Kp x Np)
__global__ void pad_w_kernel(const float* __restrict__ src, unsigned short* __restrict__ dst,
                             int K, int N, int Kp, int Np) {
  int idx = blockIdx.x * blockDim.x + threadIdx.x;
  if (idx >= Kp * Np) return;
  int r = idx / Np, c = idx % Np;
  float v = (r < K && c < N) ? src[r * N + c] : 0.0f;
  dst[idx] = f2bf(v);
}

__global__ void pad_b_kernel(const float* __restrict__ src, float* __restrict__ dst, int N, int Np) {
  int c = blockIdx.x * blockDim.x + threadIdx.x;
  if (c >= Np) return;
  dst[c] = (c < N) ? src[c] : 0.0f;
}

// x (1024x16 f32) -> A0 (1024x32 bf16) keeping first 6 features, rest zero
__global__ void build_a0_kernel(const float* __restrict__ x, unsigned short* __restrict__ A0) {
  int idx = blockIdx.x * blockDim.x + threadIdx.x;
  if (idx >= MNODE * 32) return;
  int r = idx >> 5, c = idx & 31;
  float v = (c < 6) ? x[r * 16 + c] : 0.0f;
  A0[idx] = f2bf(v);
}

// per-graph column sums of y (1024x256 f32) -> S (4x256)
__global__ void colsum_kernel(const float* __restrict__ y, float* __restrict__ S) {
  int g = blockIdx.x, c = threadIdx.x;
  float s = 0.0f;
  for (int r = 0; r < NN; ++r) s += y[(size_t)(g * NN + r) * HH + c];
  S[g * HH + c] = s;
}

// GCN epilogue: v = (S_g - y)/255 + b ; mode 0: leaky -> f32+bf16
//               mode 1: leaky + prev residual -> bf16 ; mode 2: linear -> f32(d_out)+bf16
__global__ void gcn_epi_kernel(const float* __restrict__ y, const float* __restrict__ S,
                               const float* __restrict__ b, const float* __restrict__ prev,
                               float* __restrict__ outF, unsigned short* __restrict__ outB,
                               int mode) {
  int idx = blockIdx.x * blockDim.x + threadIdx.x;   // 1024*256 total
  int c = idx & 255, row = idx >> 8, g = row >> 8;
  float v = (S[g * HH + c] - y[idx]) * COEF + b[c];
  if (mode != 2) v = (v >= 0.0f) ? v : SLOPE * v;
  if (mode == 1) v += prev[idx];
  if (outF) outF[idx] = v;
  if (outB) outB[idx] = f2bf(v);
}

// Scatter 4x4 blocks into X and S (each thread does one (g, a, b) block)
__global__ void build_xs_kernel(const float* __restrict__ vp, const float* __restrict__ vd,
                                const float* __restrict__ ep, const float* __restrict__ ed,
                                const int* __restrict__ emap,
                                float* __restrict__ X, float* __restrict__ S) {
  const int SM[16] = {0, 1, 2, 3, 1, 4, 5, 6, 2, 5, 7, 8, 3, 6, 8, 9};
  int idx = blockIdx.x * blockDim.x + threadIdx.x;
  if (idx >= GG * NN * NN) return;
  int g = idx >> 16, rem = idx & 65535;
  int a = rem >> 8, b = rem & 255;

  float px[10], dx[16];
  if (a == b) {
    const float* vpr = vp + (size_t)(g * NN + a) * 16;
    const float* vdr = vd + (size_t)(g * NN + a) * 16;
    float dv[10];
#pragma unroll
    for (int i = 0; i < 10; ++i) { px[i] = vpr[i]; dv[i] = vdr[i]; }
#pragma unroll
    for (int i = 0; i < 16; ++i) dx[i] = dv[SM[i]];
  } else {
    int e = emap[a * NN + b];
    const float* epr = ep + (size_t)(g * EUD + e) * 16;
    const float* edr = ed + (size_t)(g * EUD + e) * 16;
#pragma unroll
    for (int i = 0; i < 10; ++i) px[i] = epr[i];
    if (a < b) {
#pragma unroll
      for (int i = 0; i < 16; ++i) dx[i] = edr[i];
    } else {
#pragma unroll
      for (int i = 0; i < 4; ++i)
#pragma unroll
        for (int j = 0; j < 4; ++j) dx[i * 4 + j] = edr[j * 4 + i];
    }
  }
  float xx[16];
#pragma unroll
  for (int i = 0; i < 16; ++i) xx[i] = px[SM[i]];

  float* Xg = X + (size_t)g * 1024 * 1024;
  float* Sg = S + (size_t)g * 1024 * 1024;
#pragma unroll
  for (int i = 0; i < 4; ++i)
#pragma unroll
    for (int j = 0; j < 4; ++j) {
      size_t off = (size_t)(a * 4 + i) * 1024 + (b * 4 + j);
      Xg[off] = xx[i * 4 + j];
      Sg[off] = dx[i * 4 + j];
    }
}

// ---------------------------------------------------------------------------
extern "C" void kernel_launch(void* const* d_in, const int* in_sizes, int n_in,
                              void* d_out, int out_size, void* d_ws, size_t ws_size,
                              hipStream_t stream) {
  (void)in_sizes; (void)n_in; (void)out_size; (void)ws_size;

  const float* x = (const float*)d_in[0];
  const float* gw[3] = {(const float*)d_in[1], (const float*)d_in[3], (const float*)d_in[5]};
  const float* gb[3] = {(const float*)d_in[2], (const float*)d_in[4], (const float*)d_in[6]};
  const float* mw[4][3]; const float* mb[4][3];   // m: 0=pn 1=dn 2=pe 3=de
  for (int m = 0; m < 4; ++m)
    for (int l = 0; l < 3; ++l) {
      mw[m][l] = (const float*)d_in[7 + m * 6 + l * 2];
      mb[m][l] = (const float*)d_in[8 + m * 6 + l * 2];
    }
  const int* ud   = (const int*)d_in[32];
  const int* emap = (const int*)d_in[33];

  float* xout = (float*)d_out;                          // 1024*256
  float* Xmat = xout + (size_t)MNODE * HH;              // 4*1024*1024
  float* Smat = Xmat + (size_t)GG * 1024 * 1024;        // 4*1024*1024

  // ---- workspace bump allocator (~54 MB total) ----
  size_t off = 0;
  auto alloc = [&](size_t bytes) -> void* {
    void* p = (char*)d_ws + off;
    off += (bytes + 255) & ~(size_t)255;
    return p;
  };
  unsigned short* Wg[3];
  Wg[0] = (unsigned short*)alloc(32 * 256 * 2);
  Wg[1] = (unsigned short*)alloc(256 * 256 * 2);
  Wg[2] = (unsigned short*)alloc(256 * 256 * 2);
  unsigned short* Wm[4][3]; float* b2p[4];
  for (int m = 0; m < 4; ++m) {
    Wm[m][0] = (unsigned short*)alloc(256 * 256 * 2);
    Wm[m][1] = (unsigned short*)alloc(256 * 256 * 2);
    Wm[m][2] = (unsigned short*)alloc(256 * 16 * 2);
    b2p[m]   = (float*)alloc(16 * 4);
  }
  unsigned short* A0  = (unsigned short*)alloc((size_t)MNODE * 32 * 2);
  float* ybuf = (float*)alloc((size_t)MNODE * HH * 4);
  float* Ssum = (float*)alloc((size_t)GG * HH * 4);
  float* hF   = (float*)alloc((size_t)MNODE * HH * 4);
  unsigned short* hB  = (unsigned short*)alloc((size_t)MNODE * HH * 2);
  unsigned short* h2B = (unsigned short*)alloc((size_t)MNODE * HH * 2);
  unsigned short* xgB = (unsigned short*)alloc((size_t)MNODE * HH * 2);
  unsigned short* t1  = (unsigned short*)alloc((size_t)MNODE * HH * 2);
  unsigned short* t2  = (unsigned short*)alloc((size_t)MNODE * HH * 2);
  float* vp = (float*)alloc((size_t)MNODE * 16 * 4);
  float* vd = (float*)alloc((size_t)MNODE * 16 * 4);
  unsigned short* e1 = (unsigned short*)alloc((size_t)EUD * HH * 2);
  unsigned short* e2 = (unsigned short*)alloc((size_t)EUD * HH * 2);
  float* ep = (float*)alloc((size_t)GG * EUD * 16 * 4);
  float* ed = (float*)alloc((size_t)GG * EUD * 16 * 4);

  auto padw = [&](const float* s, unsigned short* d, int K, int N, int Kp, int Np) {
    int tot = Kp * Np;
    pad_w_kernel<<<(tot + 255) / 256, 256, 0, stream>>>(s, d, K, N, Kp, Np);
  };
  padw(gw[0], Wg[0], 6, 256, 32, 256);
  padw(gw[1], Wg[1], 256, 256, 256, 256);
  padw(gw[2], Wg[2], 256, 256, 256, 256);
  const int outdim[4] = {10, 10, 10, 16};
  for (int m = 0; m < 4; ++m) {
    padw(mw[m][0], Wm[m][0], 256, 256, 256, 256);
    padw(mw[m][1], Wm[m][1], 256, 256, 256, 256);
    padw(mw[m][2], Wm[m][2], 256, outdim[m], 256, 16);
    pad_b_kernel<<<1, 16, 0, stream>>>(mb[m][2], b2p[m], outdim[m], 16);
  }
  build_a0_kernel<<<(MNODE * 32 + 255) / 256, 256, 0, stream>>>(x, A0);

  auto gemm = [&](const unsigned short* A, const unsigned short* Xp, int nbase,
                  const unsigned short* B, const float* bias, float* Cf, unsigned short* Cb,
                  int M, int Nc, int K, int gather, int lk) {
    dim3 grid((Nc + BN - 1) / BN, (M + BM - 1) / BM);
    gemm_bf16<<<grid, 256, 0, stream>>>(A, Xp, gather ? ud : nullptr, nbase, B, bias,
                                        Cf, Cb, M, Nc, K, gather, lk);
  };

  // ---- GCN (complete graph => group-sum minus self, coef = 1/255) ----
  gemm(A0, nullptr, 0, Wg[0], nullptr, ybuf, nullptr, MNODE, 256, 32, 0, 0);
  colsum_kernel<<<GG, 256, 0, stream>>>(ybuf, Ssum);
  gcn_epi_kernel<<<MNODE, 256, 0, stream>>>(ybuf, Ssum, gb[0], nullptr, hF, hB, 0);

  gemm(hB, nullptr, 0, Wg[1], nullptr, ybuf, nullptr, MNODE, 256, 256, 0, 0);
  colsum_kernel<<<GG, 256, 0, stream>>>(ybuf, Ssum);
  gcn_epi_kernel<<<MNODE, 256, 0, stream>>>(ybuf, Ssum, gb[1], hF, nullptr, h2B, 1);

  gemm(h2B, nullptr, 0, Wg[2], nullptr, ybuf, nullptr, MNODE, 256, 256, 0, 0);
  colsum_kernel<<<GG, 256, 0, stream>>>(ybuf, Ssum);
  gcn_epi_kernel<<<MNODE, 256, 0, stream>>>(ybuf, Ssum, gb[2], nullptr, xout, xgB, 2);

  // ---- node MLPs: pn -> vp, dn -> vd ----
  float* vouts[2] = {vp, vd};
  for (int m = 0; m < 2; ++m) {
    gemm(xgB, nullptr, 0, Wm[m][0], mb[m][0], nullptr, t1, MNODE, 256, 256, 0, 1);
    gemm(t1,  nullptr, 0, Wm[m][1], mb[m][1], nullptr, t2, MNODE, 256, 256, 0, 1);
    gemm(t2,  nullptr, 0, Wm[m][2], b2p[m],   vouts[m], nullptr, MNODE, 16, 256, 0, 0);
  }

  // ---- edge MLPs (per graph; A rows gathered as xg[u]+xg[v] on the fly) ----
  for (int g = 0; g < GG; ++g) {
    for (int m = 2; m < 4; ++m) {
      float* eout = (m == 2 ? ep : ed) + (size_t)g * EUD * 16;
      gemm(nullptr, xgB, g * NN, Wm[m][0], mb[m][0], nullptr, e1, EUD, 256, 256, 1, 1);
      gemm(e1, nullptr, 0, Wm[m][1], mb[m][1], nullptr, e2, EUD, 256, 256, 0, 1);
      gemm(e2, nullptr, 0, Wm[m][2], b2p[m], eout, nullptr, EUD, 16, 256, 0, 0);
    }
  }

  // ---- assemble X and S ----
  build_xs_kernel<<<(GG * NN * NN + 255) / 256, 256, 0, stream>>>(vp, vd, ep, ed, emap, Xmat, Smat);
}
